// SparseWindowAttention_7593502179710
// MI455X (gfx1250) — compile-verified, matrix operand order checked by reading, not run on hardware
//
#include <hip/hip_runtime.h>
#include <hip/hip_bf16.h>

#define DIM 512
#define HEADS 8
#define HD 64
#define NPTS 32768
#define NW 512
#define QKV_COLS 1536
#define KSTEPS 16  // 512 / 32

typedef __bf16 bf16_t;
typedef __attribute__((ext_vector_type(16))) __bf16 v16bf;
typedef __attribute__((ext_vector_type(8)))  __bf16 v8bf;
typedef __attribute__((ext_vector_type(8)))  float  v8f;

// WMMA fragment slice from LDS: p = rowbase + 8*h; elements [0..7] = K 8h..8h+7,
// [8..15] = K 8h+16..8h+23 (per CDNA5 16-bit A/B VGPR layout, wave32).
__device__ __forceinline__ v16bf ld_frag(const bf16_t* p) {
  v16bf f;
#pragma unroll
  for (int e = 0; e < 8; ++e) { f[e] = p[e]; f[e + 8] = p[e + 16]; }
  return f;
}

__device__ __forceinline__ v8f wmma_bf16(v16bf a, v16bf b, v8f c) {
  return __builtin_amdgcn_wmma_f32_16x16x32_bf16(false, a, false, b, (short)0, c,
                                                 false, false);
}

__device__ __forceinline__ uint32_t lds_off(const void* p) {
  return (uint32_t)(uintptr_t)p;  // low 32 bits of flat LDS pointer = LDS offset
}

// one async 16B global->LDS copy, per-lane addresses, tracked by ASYNCcnt
__device__ __forceinline__ void async_cp16(uint32_t lds, uint64_t gaddr) {
  asm volatile("global_load_async_to_lds_b128 %0, %1, off"
               :: "v"(lds), "v"(gaddr) : "memory");
}
__device__ __forceinline__ void async_wait0() {
  asm volatile("s_wait_asynccnt 0x0" ::: "memory");
}
__device__ __forceinline__ void async_wait3() {
  asm volatile("s_wait_asynccnt 0x3" ::: "memory");
}

// ---------------------------------------------------------------------------
// 1) counting sort by window id (each 4^3 window holds exactly 64 points)
// ---------------------------------------------------------------------------
__global__ __launch_bounds__(256) void k_order(const int* __restrict__ coords,
                                               int* __restrict__ counts,
                                               int* __restrict__ order) {
  int i = blockIdx.x * 256 + threadIdx.x;
  int cx = coords[3 * i + 0] >> 2;
  int cy = coords[3 * i + 1] >> 2;
  int cz = coords[3 * i + 2] >> 2;
  int wid = (cx * 8 + cy) * 8 + cz;
  int slot = atomicAdd(&counts[wid], 1);
  order[wid * 64 + slot] = i;
}

// ---------------------------------------------------------------------------
// 2a) weight convert + transpose to bf16 column-major: wt[n][k] = bf16(w[k][n])
// ---------------------------------------------------------------------------
__global__ __launch_bounds__(256) void k_wT(const float* __restrict__ w,
                                            bf16_t* __restrict__ wt, int ncols) {
  int idx = blockIdx.x * 256 + threadIdx.x;  // over k * ncols, coalesced read
  int k = idx / ncols, n = idx % ncols;
  wt[(size_t)n * DIM + k] = (bf16_t)w[idx];
}

// ---------------------------------------------------------------------------
// 2b) gather + convert: xs[i][:] = bf16(x[order[i]][:])
// ---------------------------------------------------------------------------
__global__ __launch_bounds__(256) void k_gather(const float* __restrict__ x,
                                                const int* __restrict__ order,
                                                bf16_t* __restrict__ xs) {
  int u = blockIdx.x * 256 + threadIdx.x;  // unit = 4 floats
  int row = u >> 7;                        // 128 units per 512-wide row
  int c = (u & 127) * 4;
  int src = order[row];
  const float4 v = *(const float4*)(x + (size_t)src * DIM + c);
  bf16_t* o = xs + (size_t)row * DIM + c;
  o[0] = (bf16_t)v.x; o[1] = (bf16_t)v.y; o[2] = (bf16_t)v.z; o[3] = (bf16_t)v.w;
}

// ---------------------------------------------------------------------------
// 3) QKV GEMM: qkv = xs @ w_qkv + b  (A bf16 row-major, B bf16 col-major)
//    64x128 block tile, 8 waves 4x2, double-buffered async global->LDS tiles
// ---------------------------------------------------------------------------
__global__ __launch_bounds__(256) void k_gemm_qkv(
    const bf16_t* __restrict__ xs, const bf16_t* __restrict__ wt,
    const float* __restrict__ b, bf16_t* __restrict__ qkvb) {
  __shared__ alignas(16) bf16_t sA[2][64 * 32];
  __shared__ alignas(16) bf16_t sB[2][128 * 32];
  const int t = threadIdx.x;
  const int bx = blockIdx.x, by = blockIdx.y;
  const int wave = t >> 5, lane = t & 31;
  const int wr = wave >> 1, wc = wave & 1;
  const int mn = lane & 15, h = lane >> 4;

  // async copy plan: A tile 64x32 = 256 x 16B (1/thread); B tile 128x32 (2/thread)
  const int ar = t >> 2, ac = (t & 3) * 8;
  const int bn = t >> 2;
  uint64_t gA = (uint64_t)(uintptr_t)(xs + (size_t)(by * 64 + ar) * DIM + ac);
  uint64_t gB0 = (uint64_t)(uintptr_t)(wt + (size_t)(bx * 128 + bn) * DIM + ac);
  uint64_t gB1 = (uint64_t)(uintptr_t)(wt + (size_t)(bx * 128 + bn + 64) * DIM + ac);
  uint32_t lA[2], lB0[2], lB1[2];
#pragma unroll
  for (int s = 0; s < 2; ++s) {
    lA[s] = lds_off(&sA[s][ar * 32 + ac]);
    lB0[s] = lds_off(&sB[s][bn * 32 + ac]);
    lB1[s] = lds_off(&sB[s][(bn + 64) * 32 + ac]);
  }

  // prologue: tile 0 into buffer 0
  async_cp16(lA[0], gA); async_cp16(lB0[0], gB0); async_cp16(lB1[0], gB1);
  gA += 64; gB0 += 64; gB1 += 64;  // advance 32 bf16 = 64 B

  v8f acc[4] = {};
  for (int s = 0; s < KSTEPS; ++s) {
    int cur = s & 1;
    if (s + 1 < KSTEPS) {  // issue next tile into other buffer
      int nxt = cur ^ 1;
      async_cp16(lA[nxt], gA); async_cp16(lB0[nxt], gB0); async_cp16(lB1[nxt], gB1);
      gA += 64; gB0 += 64; gB1 += 64;
      async_wait3();  // retire current tile (in-order), keep next in flight
    } else {
      async_wait0();
    }
    __syncthreads();
    v16bf a = ld_frag(&sA[cur][(wr * 16 + mn) * 32 + 8 * h]);
#pragma unroll
    for (int j = 0; j < 4; ++j) {
      v16bf bb = ld_frag(&sB[cur][(wc * 64 + j * 16 + mn) * 32 + 8 * h]);
      acc[j] = wmma_bf16(a, bb, acc[j]);
    }
    __syncthreads();  // before buffer `cur` is re-filled at step s+1
  }
#pragma unroll
  for (int j = 0; j < 4; ++j) {
    int col = bx * 128 + wc * 64 + j * 16 + mn;
    float bias = b[col];
#pragma unroll
    for (int r = 0; r < 8; ++r) {
      int m = wr * 16 + r + 8 * h;
      qkvb[(size_t)(by * 64 + m) * QKV_COLS + col] = (bf16_t)(acc[j][r] + bias);
    }
  }
}

// ---------------------------------------------------------------------------
// 4) windowed attention: one block per (head, window); 64x64 scores
// ---------------------------------------------------------------------------
__global__ __launch_bounds__(256) void k_attn(
    const bf16_t* __restrict__ qkvb, const int* __restrict__ order,
    const int* __restrict__ coords, const float* __restrict__ rpb,
    bf16_t* __restrict__ attnb) {
  __shared__ alignas(16) bf16_t sQ[64 * 64];
  __shared__ alignas(16) bf16_t sK[64 * 64];   // row-major == B-frag layout for QK^T
  __shared__ alignas(16) bf16_t sVt[64 * 64];  // V transposed: [d][p]
  __shared__ alignas(16) float  sS[64 * 64];
  __shared__ alignas(16) bf16_t sP[64 * 64];
  __shared__ float sRed[256];
  __shared__ int sC[64 * 3];

  const int t = threadIdx.x;
  const int head = blockIdx.x;
  const int g = blockIdx.y * 64;
  const int wave = t >> 5, lane = t & 31;
  const int wr = wave >> 1, wc = wave & 1;
  const int mn = lane & 15, h = lane >> 4;

  // Q,K tiles: async 16B copies; V: load + transposed store
  for (int u = t; u < 512; u += 256) {
    int p = u >> 3, c = (u & 7) * 8;
    const bf16_t* base = qkvb + (size_t)(g + p) * QKV_COLS + head * HD + c;
    async_cp16(lds_off(&sQ[p * 64 + c]), (uint64_t)(uintptr_t)base);
    async_cp16(lds_off(&sK[p * 64 + c]), (uint64_t)(uintptr_t)(base + DIM));
    v8bf vv = *(const v8bf*)(base + 2 * DIM);
#pragma unroll
    for (int j = 0; j < 8; ++j) sVt[(c + j) * 64 + p] = vv[j];
  }
  if (t < 64) {
    int src = order[g + t];
    sC[t * 3 + 0] = coords[src * 3 + 0];
    sC[t * 3 + 1] = coords[src * 3 + 1];
    sC[t * 3 + 2] = coords[src * 3 + 2];
  }
  async_wait0();
  __syncthreads();

  // S = (Q K^T) * scale + rel_pos_bias
#pragma unroll
  for (int j = 0; j < 2; ++j) {
    int nt = wc * 2 + j;
    v8f acc = {};
#pragma unroll
    for (int ks = 0; ks < 2; ++ks) {
      v16bf a  = ld_frag(&sQ[(wr * 16 + mn) * 64 + ks * 32 + 8 * h]);
      v16bf bb = ld_frag(&sK[(nt * 16 + mn) * 64 + ks * 32 + 8 * h]);
      acc = wmma_bf16(a, bb, acc);
    }
    int n = nt * 16 + mn;
#pragma unroll
    for (int r = 0; r < 8; ++r) {
      int m = wr * 16 + r + 8 * h;
      int rx = sC[m * 3 + 0] - sC[n * 3 + 0] + 3;
      int ry = sC[m * 3 + 1] - sC[n * 3 + 1] + 3;
      int rz = sC[m * 3 + 2] - sC[n * 3 + 2] + 3;
      int ridx = (rx * 7 + ry) * 7 + rz;
      sS[m * 64 + n] = acc[r] * 0.125f + rpb[ridx * HEADS + head];
    }
  }
  __syncthreads();

  // softmax: 4 threads per row, LDS reduction
  {
    int row = t >> 2, q4 = (t & 3) * 16;
    float mx = -3.0e38f;
    for (int n = q4; n < q4 + 16; ++n) mx = fmaxf(mx, sS[row * 64 + n]);
    sRed[t] = mx;
    __syncthreads();
    float rmx = fmaxf(fmaxf(sRed[row * 4 + 0], sRed[row * 4 + 1]),
                      fmaxf(sRed[row * 4 + 2], sRed[row * 4 + 3]));
    __syncthreads();
    float sum = 0.f;
    for (int n = q4; n < q4 + 16; ++n) {
      float e = __expf(sS[row * 64 + n] - rmx);
      sS[row * 64 + n] = e;
      sum += e;
    }
    sRed[t] = sum;
    __syncthreads();
    float inv = 1.0f / (sRed[row * 4 + 0] + sRed[row * 4 + 1] +
                        sRed[row * 4 + 2] + sRed[row * 4 + 3]);
    for (int n = q4; n < q4 + 16; ++n)
      sP[row * 64 + n] = (bf16_t)(sS[row * 64 + n] * inv);
  }
  __syncthreads();

  // O = P @ V
#pragma unroll
  for (int j = 0; j < 2; ++j) {
    int nt = wc * 2 + j;
    v8f acc = {};
#pragma unroll
    for (int ks = 0; ks < 2; ++ks) {
      v16bf a  = ld_frag(&sP[(wr * 16 + mn) * 64 + ks * 32 + 8 * h]);
      v16bf bb = ld_frag(&sVt[(nt * 16 + mn) * 64 + ks * 32 + 8 * h]);
      acc = wmma_bf16(a, bb, acc);
    }
    int d = nt * 16 + mn;
#pragma unroll
    for (int r = 0; r < 8; ++r) {
      int m = wr * 16 + r + 8 * h;
      attnb[(size_t)(g + m) * DIM + head * HD + d] = (bf16_t)acc[r];
    }
  }
}

// ---------------------------------------------------------------------------
// 5) projection GEMM + scatter: out[order[i],:] = attn[i,:] @ w_proj + b_proj
// ---------------------------------------------------------------------------
__global__ __launch_bounds__(256) void k_gemm_proj(
    const bf16_t* __restrict__ attnb, const bf16_t* __restrict__ wt,
    const float* __restrict__ b, const int* __restrict__ order,
    float* __restrict__ out) {
  __shared__ alignas(16) bf16_t sA[2][64 * 32];
  __shared__ alignas(16) bf16_t sB[2][128 * 32];
  __shared__ int rows[64];
  const int t = threadIdx.x;
  const int bx = blockIdx.x, by = blockIdx.y;
  const int wave = t >> 5, lane = t & 31;
  const int wr = wave >> 1, wc = wave & 1;
  const int mn = lane & 15, h = lane >> 4;

  if (t < 64) rows[t] = order[by * 64 + t];

  const int ar = t >> 2, ac = (t & 3) * 8;
  const int bn = t >> 2;
  uint64_t gA = (uint64_t)(uintptr_t)(attnb + (size_t)(by * 64 + ar) * DIM + ac);
  uint64_t gB0 = (uint64_t)(uintptr_t)(wt + (size_t)(bx * 128 + bn) * DIM + ac);
  uint64_t gB1 = (uint64_t)(uintptr_t)(wt + (size_t)(bx * 128 + bn + 64) * DIM + ac);
  uint32_t lA[2], lB0[2], lB1[2];
#pragma unroll
  for (int s = 0; s < 2; ++s) {
    lA[s] = lds_off(&sA[s][ar * 32 + ac]);
    lB0[s] = lds_off(&sB[s][bn * 32 + ac]);
    lB1[s] = lds_off(&sB[s][(bn + 64) * 32 + ac]);
  }
  async_cp16(lA[0], gA); async_cp16(lB0[0], gB0); async_cp16(lB1[0], gB1);
  gA += 64; gB0 += 64; gB1 += 64;

  v8f acc[4] = {};
  for (int s = 0; s < KSTEPS; ++s) {
    int cur = s & 1;
    if (s + 1 < KSTEPS) {
      int nxt = cur ^ 1;
      async_cp16(lA[nxt], gA); async_cp16(lB0[nxt], gB0); async_cp16(lB1[nxt], gB1);
      gA += 64; gB0 += 64; gB1 += 64;
      async_wait3();
    } else {
      async_wait0();
    }
    __syncthreads();
    v16bf a = ld_frag(&sA[cur][(wr * 16 + mn) * 32 + 8 * h]);
#pragma unroll
    for (int j = 0; j < 4; ++j) {
      v16bf bb = ld_frag(&sB[cur][(wc * 64 + j * 16 + mn) * 32 + 8 * h]);
      acc[j] = wmma_bf16(a, bb, acc[j]);
    }
    __syncthreads();
  }
#pragma unroll
  for (int j = 0; j < 4; ++j) {
    int col = bx * 128 + wc * 64 + j * 16 + mn;
    float bias = b[col];
#pragma unroll
    for (int r = 0; r < 8; ++r) {
      int m = wr * 16 + r + 8 * h;
      out[(size_t)rows[m] * DIM + col] = acc[j][r] + bias;
    }
  }
}

// ---------------------------------------------------------------------------
extern "C" void kernel_launch(void* const* d_in, const int* in_sizes, int n_in,
                              void* d_out, int out_size, void* d_ws,
                              size_t ws_size, hipStream_t stream) {
  (void)in_sizes; (void)n_in; (void)out_size; (void)ws_size;
  const float* x      = (const float*)d_in[0];
  const int*   coords = (const int*)d_in[1];
  const float* w_qkv  = (const float*)d_in[2];
  const float* b_qkv  = (const float*)d_in[3];
  const float* w_proj = (const float*)d_in[4];
  const float* b_proj = (const float*)d_in[5];
  const float* rpb    = (const float*)d_in[6];
  float* out = (float*)d_out;

  char* ws = (char*)d_ws;
  int* counts = (int*)ws;                                    // @0       2 KB
  int* order  = (int*)(ws + 2048);                           // @2KB   128 KB
  bf16_t* xs    = (bf16_t*)(ws + 133120);                    //         32 MB
  bf16_t* wtq   = (bf16_t*)(ws + 133120 + 33554432);         //        1.5 MB
  bf16_t* wtp   = (bf16_t*)(ws + 133120 + 33554432 + 1572864);        // 0.5 MB
  bf16_t* qkvb  = (bf16_t*)(ws + 133120 + 33554432 + 1572864 + 524288); // 96 MB
  bf16_t* attnb = (bf16_t*)(ws + 133120 + 33554432 + 1572864 + 524288 +
                            100663296);                      //         32 MB

  hipMemsetAsync(counts, 0, 512 * sizeof(int), stream);
  k_order<<<NPTS / 256, 256, 0, stream>>>(coords, counts, order);
  k_wT<<<(DIM * QKV_COLS) / 256, 256, 0, stream>>>(w_qkv, wtq, QKV_COLS);
  k_wT<<<(DIM * DIM) / 256, 256, 0, stream>>>(w_proj, wtp, DIM);
  k_gather<<<(NPTS * 128) / 256, 256, 0, stream>>>(x, order, xs);
  k_gemm_qkv<<<dim3(QKV_COLS / 128, NPTS / 64), 256, 0, stream>>>(xs, wtq, b_qkv,
                                                                  qkvb);
  k_attn<<<dim3(HEADS, NW), 256, 0, stream>>>(qkvb, order, coords, rpb, attnb);
  k_gemm_proj<<<dim3(DIM / 128, NPTS / 64), 256, 0, stream>>>(attnb, wtp, b_proj,
                                                              order, out);
}
